// PQHDC_38036230373923
// MI455X (gfx1250) — 8-bit matrix operand order repaired and checked by reading, compiled, not run on hardware
//
#include <hip/hip_runtime.h>
#include <stdint.h>

typedef __attribute__((ext_vector_type(8))) int v8i;

#define D_REAL 10000
#define D_PAD  10048
#define KT     157
#define B_ROWS 4096
#define C_CLS  128
#define N_COLS 384
#define NT     24
#define MT_TOT 256

#define QPACK_BYTES ((size_t)MT_TOT * KT * 1024)
#define BPACK_BYTES ((size_t)NT * KT * 1024)

__global__ __launch_bounds__(256) void pq_zero_u32(uint32_t* p, int n) {
    int i = blockIdx.x * 256 + threadIdx.x;
    if (i < n) p[i] = 0u;
}

__global__ __launch_bounds__(256) void pq_pack_q(const float* __restrict__ q,
                                                 uint8_t* __restrict__ qpack,
                                                 uint32_t* __restrict__ qsum) {
    int gwave = (blockIdx.x * 256 + threadIdx.x) >> 5;
    int lane  = threadIdx.x & 31;
    int mt = gwave / KT;
    int kt = gwave - mt * KT;
    int m  = mt * 16 + (lane & 15);
    int hi = lane >> 4;

    const float* row = q + (size_t)m * D_REAL;
    uint32_t dw[8];
    uint32_t cnt = 0;

    if (kt < KT - 1) {
        #pragma unroll
        for (int v = 0; v < 8; ++v) {
            int kbase = kt * 64 + ((v >> 1) << 4) + ((v & 1) << 2) + (hi << 3);
            float4 f = *(const float4*)(row + kbase);
            uint32_t b0 = f.x > 0.f, b1 = f.y > 0.f, b2 = f.z > 0.f, b3 = f.w > 0.f;
            dw[v] = b0 | (b1 << 8) | (b2 << 16) | (b3 << 24);
            cnt += b0 + b1 + b2 + b3;
        }
    } else {
        #pragma unroll
        for (int v = 0; v < 8; ++v) {
            int kbase = kt * 64 + ((v >> 1) << 4) + ((v & 1) << 2) + (hi << 3);
            uint32_t d = 0;
            #pragma unroll
            for (int j = 0; j < 4; ++j) {
                int k = kbase + j;
                uint32_t bit = (k < D_REAL && row[k] > 0.f) ? 1u : 0u;
                d |= bit << (8 * j);
                cnt += bit;
            }
            dw[v] = d;
        }
    }

    uint8_t* dst = qpack + (size_t)gwave * 1024 + lane * 32;
    *(int4*)(dst)      = make_int4((int)dw[0], (int)dw[1], (int)dw[2], (int)dw[3]);
    *(int4*)(dst + 16) = make_int4((int)dw[4], (int)dw[5], (int)dw[6], (int)dw[7]);
    atomicAdd(&qsum[m], cnt);
}

__device__ __forceinline__ float pq_block_reduce(float v, float* red) {
    int t = threadIdx.x;
    red[t] = v;
    __syncthreads();
    #pragma unroll
    for (int s = 128; s > 0; s >>= 1) {
        if (t < s) red[t] += red[t + s];
        __syncthreads();
    }
    float r = red[0];
    __syncthreads();
    return r;
}

__global__ __launch_bounds__(256) void pq_transform(const float* __restrict__ am,
                                                    uint8_t* __restrict__ bpack,
                                                    float* __restrict__ wfin,
                                                    float* __restrict__ biasv,
                                                    float* __restrict__ wsumv) {
    __shared__ float red[256];
    int c   = blockIdx.x;
    int tid = threadIdx.x;
    const float* hv = am + (size_t)c * D_REAL;
    const float Df = (float)D_REAL;

    float sabs = 0.f, a1 = 0.f;
    for (int d = tid; d < D_REAL; d += 256) {
        float h = hv[d];
        sabs += fabsf(h);
        a1 += (h > 0.f) ? 1.f : 0.f;
    }
    float s1    = pq_block_reduce(sabs, red);
    float asum1 = pq_block_reduce(a1, red);
    float m1 = s1 / Df;

    float sabs2 = 0.f, y2 = 0.f, g12 = 0.f, a2 = 0.f;
    for (int d = tid; d < D_REAL; d += 256) {
        float h  = hv[d];
        float x1 = (h > 0.f) ? 1.f : -1.f;
        float l2 = h - x1 * m1;
        float x2 = (l2 > 0.f) ? 1.f : -1.f;
        sabs2 += fabsf(l2);
        y2  += h * x2;
        g12 += x1 * x2;
        a2  += (l2 > 0.f) ? 1.f : 0.f;
    }
    float s2    = pq_block_reduce(sabs2, red);
    float Y2    = pq_block_reduce(y2, red);
    float G12   = pq_block_reduce(g12, red);
    float asum2 = pq_block_reduce(a2, red);
    float m2 = s2 / Df;

    float y3 = 0.f, g13 = 0.f, g23 = 0.f, a3 = 0.f;
    for (int d = tid; d < D_PAD; d += 256) {
        uint8_t vals[3] = {0, 0, 0};
        if (d < D_REAL) {
            float h  = hv[d];
            float x1 = (h > 0.f) ? 1.f : -1.f;
            float l2 = h - x1 * m1;
            float x2 = (l2 > 0.f) ? 1.f : -1.f;
            float l3 = l2 - x2 * m2;
            float x3 = (l3 > 0.f) ? 1.f : -1.f;
            y3  += h * x3;
            g13 += x1 * x3;
            g23 += x2 * x3;
            a3  += (l3 > 0.f) ? 1.f : 0.f;
            vals[0] = (h  > 0.f);
            vals[1] = (l2 > 0.f);
            vals[2] = (l3 > 0.f);
        }
        int kt = d >> 6, r = d & 63;
        int j  = r & 3;
        int v  = ((r >> 4) << 1) + ((r >> 2) & 1);
        int hi = (r >> 3) & 1;
        #pragma unroll
        for (int p = 0; p < 3; ++p) {
            int n    = c * 3 + p;
            int lane = (n & 15) + (hi << 4);
            bpack[((size_t)(n >> 4) * KT + kt) * 1024 + lane * 32 + v * 4 + j] = vals[p];
        }
    }
    float Y3    = pq_block_reduce(y3, red);
    float G13   = pq_block_reduce(g13, red);
    float G23   = pq_block_reduce(g23, red);
    float asum3 = pq_block_reduce(a3, red);

    if (tid == 0) {
        float Y1 = s1;
        float c11 = Df * Df - G23 * G23;
        float c12 = G13 * G23 - G12 * Df;
        float c13 = G12 * G23 - G13 * Df;
        float det = Df * c11 + G12 * c12 + G13 * c13;
        float i11 = c11 / det, i12 = c12 / det, i13 = c13 / det;
        float i22 = (Df * Df - G13 * G13) / det;
        float i23 = (G12 * G13 - Df * G23) / det;
        float i33 = (Df * Df - G12 * G12) / det;
        float w1 = Y1 * i11 + Y2 * i12 + Y3 * i13;
        float w2 = Y1 * i12 + Y2 * i22 + Y3 * i23;
        float w3 = Y1 * i13 + Y2 * i23 + Y3 * i33;
        float ss = w1 * w1 + w2 * w2 + w3 * w3;
        float wn0 = w1 * w1 / ss, wn1 = w2 * w2 / ss, wn2 = w3 * w3 / ss;
        wfin[c * 3 + 0] = wn0;
        wfin[c * 3 + 1] = wn1;
        wfin[c * 3 + 2] = wn2;
        biasv[c] = wn0 * (Df - asum1) + wn1 * (Df - asum2) + wn2 * (Df - asum3);
        wsumv[c] = wn0 + wn1 + wn2;
    }
}

__global__ __launch_bounds__(256) void pq_gemm(const uint8_t* __restrict__ qpack,
                                               const uint8_t* __restrict__ bpack,
                                               const uint32_t* __restrict__ qsum,
                                               const float* __restrict__ wfin,
                                               const float* __restrict__ biasv,
                                               const float* __restrict__ wsumv,
                                               float* __restrict__ out) {
    __shared__ float lds[16 * N_COLS];
    int lane = threadIdx.x & 31;
    int wave = threadIdx.x >> 5;
    int mt0  = blockIdx.x * 2;

    v8i acc[2][3] = {};
    const uint8_t* ap[2];
    const uint8_t* bp[3];
    #pragma unroll
    for (int i = 0; i < 2; ++i)
        ap[i] = qpack + (size_t)(mt0 + i) * KT * 1024 + lane * 32;
    #pragma unroll
    for (int j = 0; j < 3; ++j)
        bp[j] = bpack + (size_t)(wave * 3 + j) * KT * 1024 + lane * 32;

    #pragma unroll 2
    for (int kt = 0; kt < KT; ++kt) {
        v8i A[2], Bv[3];
        #pragma unroll
        for (int i = 0; i < 2; ++i) A[i]  = *(const v8i*)(ap[i] + (size_t)kt * 1024);
        #pragma unroll
        for (int j = 0; j < 3; ++j) Bv[j] = *(const v8i*)(bp[j] + (size_t)kt * 1024);
        #pragma unroll
        for (int i = 0; i < 2; ++i)
            #pragma unroll
            for (int j = 0; j < 3; ++j)
                acc[i][j] = __builtin_amdgcn_wmma_i32_16x16x64_iu8(
                    false, A[i], false, Bv[j], acc[i][j], false, false);
    }

    int rbase = (lane >> 4) << 3;
    int cl    = lane & 15;
    for (int i = 0; i < 2; ++i) {
        __syncthreads();
        #pragma unroll
        for (int j = 0; j < 3; ++j) {
            int col = wave * 48 + j * 16 + cl;
            #pragma unroll
            for (int r = 0; r < 8; ++r)
                lds[(rbase + r) * N_COLS + col] = (float)acc[i][j][r];
        }
        __syncthreads();
        for (int t = threadIdx.x; t < 16 * C_CLS; t += 256) {
            int r  = t >> 7;
            int cc = t & 127;
            int grow = blockIdx.x * 32 + i * 16 + r;
            const float* Lr = lds + r * N_COLS + 3 * cc;
            float dot = wfin[3 * cc] * Lr[0] + wfin[3 * cc + 1] * Lr[1]
                      + wfin[3 * cc + 2] * Lr[2];
            out[(size_t)grow * C_CLS + cc] =
                biasv[cc] - (float)qsum[grow] * wsumv[cc] + 2.f * dot;
        }
    }
}

extern "C" void kernel_launch(void* const* d_in, const int* in_sizes, int n_in,
                              void* d_out, int out_size, void* d_ws, size_t ws_size,
                              hipStream_t stream) {
    const float* query = (const float*)d_in[0];
    const float* am    = (const float*)d_in[1];
    (void)in_sizes; (void)n_in; (void)out_size; (void)ws_size;

    uint8_t*  ws    = (uint8_t*)d_ws;
    uint8_t*  qpack = ws;
    uint8_t*  bpack = ws + QPACK_BYTES;
    uint32_t* qsum  = (uint32_t*)(ws + QPACK_BYTES + BPACK_BYTES);
    float*    wfin  = (float*)((uint8_t*)qsum + B_ROWS * sizeof(uint32_t));
    float*    biasv = wfin + N_COLS;
    float*    wsumv = biasv + C_CLS;
    float*    out   = (float*)d_out;

    pq_zero_u32<<<(B_ROWS + 255) / 256, 256, 0, stream>>>(qsum, B_ROWS);
    pq_pack_q<<<(MT_TOT * KT) / 8, 256, 0, stream>>>(query, qpack, qsum);
    pq_transform<<<C_CLS, 256, 0, stream>>>(am, bpack, wfin, biasv, wsumv);
    pq_gemm<<<B_ROWS / 32, 256, 0, stream>>>(qpack, bpack, qsum, wfin, biasv, wsumv, out);
}
